// LTMLoRA_24902220383045
// MI455X (gfx1250) — matrix operand order repaired and checked by reading, not compiled, not run on hardware
//
#include <hip/hip_runtime.h>

typedef __attribute__((ext_vector_type(16))) _Float16 v16h;
typedef __attribute__((ext_vector_type(8)))  float    v8f;

constexpr int kTokens = 8192;
constexpr int kIn     = 4096;
constexpr int kOut    = 4096;
constexpr int kRank   = 64;

constexpr float kScaleU = 0.90483741803595957f;  // exp(-0.1)
constexpr float kScaleV = 1.10517091808194934f;  // exp(+0.1)

__device__ __forceinline__ v8f wmma_f16(v16h a, v16h b, v8f c) {
  // D = A(16x32 f16) * B(32x16 f16) + C(16x16 f32)
  return __builtin_amdgcn_wmma_f32_16x16x32_f16(
      /*neg_a=*/false, a, /*neg_b=*/false, b,
      /*c_mod=*/(short)0, c, /*reuse_a=*/false, /*reuse_b=*/false);
}

// Pack 16 fp32 (row `row`, K = k0+g*16+{0..15}) into the
// WMMA 16-bit A-operand lane layout (contiguous K, matching B packing):
//   lanes 0-15 (g=0): v0-7 hold K 0..15
//   lanes 16-31(g=1): v0-7 hold K 16..31
__device__ __forceinline__ v16h pack_a16(const float* __restrict__ row, int k0, int g) {
  const float4* pa = (const float4*)(row + k0 + g * 16);
  const float4* pb = (const float4*)(row + k0 + g * 16 + 8);
  float4 f0 = pa[0], f1 = pa[1], f2 = pb[0], f3 = pb[1];
  v16h a;
  a[0]  = (_Float16)f0.x; a[1]  = (_Float16)f0.y;
  a[2]  = (_Float16)f0.z; a[3]  = (_Float16)f0.w;
  a[4]  = (_Float16)f1.x; a[5]  = (_Float16)f1.y;
  a[6]  = (_Float16)f1.z; a[7]  = (_Float16)f1.w;
  a[8]  = (_Float16)f2.x; a[9]  = (_Float16)f2.y;
  a[10] = (_Float16)f2.z; a[11] = (_Float16)f2.w;
  a[12] = (_Float16)f3.x; a[13] = (_Float16)f3.y;
  a[14] = (_Float16)f3.z; a[15] = (_Float16)f3.w;
  return a;
}

// B-operand packing used here (32x16, 16-bit, contiguous K): lane = {g=K-halve, n=N column};
// element j = K row (g*16 + j). Pre-swizzle U into this layout (f16, scaled).
// Uh element index: ((kt*4 + nt)*32 + lane), kt = K/32 tile, nt = N/16 tile.
__global__ __launch_bounds__(256) void lora_pack_u(const float* __restrict__ U,
                                                   v16h* __restrict__ Uh) {
  const int t    = blockIdx.x * 256 + threadIdx.x;   // 16384 threads
  const int lane = t & 31;
  const int rest = t >> 5;
  const int nt   = rest & 3;        // 64 / 16
  const int kt   = rest >> 2;       // 4096 / 32
  const int n    = lane & 15;
  const int g    = lane >> 4;
  v16h b;
#pragma unroll
  for (int j = 0; j < 16; ++j) {
    const int k = kt * 32 + g * 16 + j;
    b[j] = (_Float16)(U[(size_t)k * kRank + nt * 16 + n] * kScaleU);
  }
  Uh[(size_t)(kt * 4 + nt) * 32 + lane] = b;
}

// Same for V: Vh element index ((kt*256 + nt)*32 + lane), kt in 0..1, nt in 0..255.
__global__ __launch_bounds__(256) void lora_pack_v(const float* __restrict__ V,
                                                   v16h* __restrict__ Vh) {
  const int t    = blockIdx.x * 256 + threadIdx.x;   // 16384 threads
  const int lane = t & 31;
  const int rest = t >> 5;
  const int nt   = rest & 255;      // 4096 / 16
  const int kt   = rest >> 8;       // 64 / 32
  const int n    = lane & 15;
  const int g    = lane >> 4;
  v16h b;
#pragma unroll
  for (int j = 0; j < 16; ++j) {
    const int k = kt * 32 + g * 16 + j;
    b[j] = (_Float16)(V[(size_t)k * kOut + nt * 16 + n] * kScaleV);
  }
  Vh[(size_t)(kt * 256 + nt) * 32 + lane] = b;
}

// Stage 1: Y[8192x64] = x @ Uh. One block per 16-row strip; 8 waves split the
// K=4096 reduction (512 each), LDS reduction combines partials.
__global__ __launch_bounds__(256) void lora_stage1(const float* __restrict__ x,
                                                   const v16h* __restrict__ Uh,
                                                   float* __restrict__ Y) {
  __shared__ float red[8 * 1024];
  const int lane = threadIdx.x & 31;
  const int wave = threadIdx.x >> 5;
  const int m0   = blockIdx.x * 16;
  const int m    = lane & 15;
  const int g    = lane >> 4;

  const float* xrow  = x + (size_t)(m0 + m) * kIn;
  const int    kBase = wave * (kIn / 8);   // 512-wide K chunk per wave

  v8f acc0 = {}, acc1 = {}, acc2 = {}, acc3 = {};

  for (int kk = 0; kk < kIn / 8; kk += 32) {
    const int k0 = kBase + kk;
    v16h a = pack_a16(xrow, k0, g);
    const int kt = k0 >> 5;
    const v16h* ub = Uh + (size_t)kt * 4 * 32 + lane;
    v16h b0 = ub[0 * 32];
    v16h b1 = ub[1 * 32];
    v16h b2 = ub[2 * 32];
    v16h b3 = ub[3 * 32];
    acc0 = wmma_f16(a, b0, acc0);
    acc1 = wmma_f16(a, b1, acc1);
    acc2 = wmma_f16(a, b2, acc2);
    acc3 = wmma_f16(a, b3, acc3);
  }

  float* slab = red + wave * 1024;
#pragma unroll
  for (int r = 0; r < 8; ++r) {
    slab[(0 * 8 + r) * 32 + lane] = acc0[r];
    slab[(1 * 8 + r) * 32 + lane] = acc1[r];
    slab[(2 * 8 + r) * 32 + lane] = acc2[r];
    slab[(3 * 8 + r) * 32 + lane] = acc3[r];
  }
  __syncthreads();

#pragma unroll
  for (int i = 0; i < 4; ++i) {
    const int idx = threadIdx.x + i * 256;   // 0..1023
    float s = 0.f;
#pragma unroll
    for (int w = 0; w < 8; ++w) s += red[w * 1024 + idx];
    const int nt = idx >> 8;
    const int r  = (idx >> 5) & 7;
    const int l  = idx & 31;
    // C/D layout: M = r + 8*(lane>=16), N = lane&15
    Y[(size_t)(m0 + r + 8 * (l >> 4)) * kRank + nt * 16 + (l & 15)] = s;
  }
}

// Stage 2: out[8192x4096] = Y @ Vh. Each wave owns a 16x128 output tile:
// K=64 -> 2 WMMA K-steps, 8 N-tiles. Nontemporal stores for the 134MB output.
__global__ __launch_bounds__(256) void lora_stage2(const float* __restrict__ Y,
                                                   const v16h* __restrict__ Vh,
                                                   float* __restrict__ out) {
  const int lane = threadIdx.x & 31;
  const int wave = threadIdx.x >> 5;
  const int job  = blockIdx.x * 8 + wave;   // 0..16383
  const int c    = job & 31;                // 128-col block
  const int m0   = (job >> 5) * 16;         // row strip
  const int m    = lane & 15;
  const int g    = lane >> 4;

  const float* yrow = Y + (size_t)(m0 + m) * kRank;
  v16h a0 = pack_a16(yrow, 0, g);
  v16h a1 = pack_a16(yrow, 32, g);

#pragma unroll
  for (int nt = 0; nt < 8; ++nt) {
    const int ntile = c * 8 + nt;           // 0..255
    v16h b0 = Vh[(size_t)(0 * 256 + ntile) * 32 + lane];
    v16h b1 = Vh[(size_t)(1 * 256 + ntile) * 32 + lane];
    v8f acc = {};
    acc = wmma_f16(a0, b0, acc);
    acc = wmma_f16(a1, b1, acc);
#pragma unroll
    for (int r = 0; r < 8; ++r) {
      __builtin_nontemporal_store(
          acc[r], out + (size_t)(m0 + r + 8 * g) * kOut + ntile * 16 + (lane & 15));
    }
  }
}

extern "C" void kernel_launch(void* const* d_in, const int* in_sizes, int n_in,
                              void* d_out, int out_size, void* d_ws, size_t ws_size,
                              hipStream_t stream) {
  (void)in_sizes; (void)n_in; (void)out_size; (void)ws_size;
  const float* x = (const float*)d_in[0];
  const float* U = (const float*)d_in[1];
  const float* V = (const float*)d_in[2];
  float* out = (float*)d_out;

  // Workspace layout: Y (2 MiB) | Uh (512 KiB) | Vh (512 KiB)  = 3 MiB
  char*  ws = (char*)d_ws;
  float* Yw = (float*)ws;
  v16h*  Uh = (v16h*)(ws + (size_t)kTokens * kRank * sizeof(float));
  v16h*  Vh = (v16h*)(ws + (size_t)kTokens * kRank * sizeof(float)
                         + (size_t)kIn * kRank * sizeof(_Float16));

  lora_pack_u<<<64, 256, 0, stream>>>(U, Uh);
  lora_pack_v<<<64, 256, 0, stream>>>(V, Vh);
  lora_stage1<<<kTokens / 16, 256, 0, stream>>>(x, Uh, Yw);
  lora_stage2<<<(kTokens / 16) * (kOut / 128) / 8, 256, 0, stream>>>(Yw, Vh, out);
}
